// L2_Self_Attn_Old_82849919140012
// MI455X (gfx1250) — matrix operand order repaired and checked by reading, not compiled, not run on hardware
//
#include <hip/hip_runtime.h>
#include <cmath>

typedef __attribute__((ext_vector_type(16))) _Float16 v16h;
typedef __attribute__((ext_vector_type(8)))  float    v8f;

#define CC 128
#define NN 4096
#define BB 4

__device__ __forceinline__ v8f wmma_f16(v16h a, v16h b, v8f c) {
  // D = A(16x32 f16) * B(32x16 f16) + C(16x16 f32)
  return __builtin_amdgcn_wmma_f32_16x16x32_f16(false, a, false, b, (short)0, c,
                                                false, false);
}

__device__ __forceinline__ float redmax16(float v) {
  v = fmaxf(v, __shfl_xor(v, 1, 32));
  v = fmaxf(v, __shfl_xor(v, 2, 32));
  v = fmaxf(v, __shfl_xor(v, 4, 32));
  v = fmaxf(v, __shfl_xor(v, 8, 32));
  return v;
}
__device__ __forceinline__ float redsum16(float v) {
  v += __shfl_xor(v, 1, 32);
  v += __shfl_xor(v, 2, 32);
  v += __shfl_xor(v, 4, 32);
  v += __shfl_xor(v, 8, 32);
  return v;
}

// ---------------------------------------------------------------------------
// Unified 1x1-conv / projection GEMM:
//   out[b, m*128+o] = scale * sum_i A[m,i] * W[o,i]  (+ bias[o])
// A view: a_rowmajor ? in[b, m*128+i]  (token-major, (N,C))
//                    : in[b, i*4096+m] (channel-major, (C,N)) -- conv1x1 view
// One block = 16 output rows (m) x all 128 cols (o); 8 waves, 1 WMMA tile each.
// ---------------------------------------------------------------------------
__global__ __launch_bounds__(256) void conv_gemm_kernel(
    const float* __restrict__ in, const float* __restrict__ W,
    const float* __restrict__ bias, float scale, float* __restrict__ out,
    int a_rowmajor) {
  __shared__ _Float16 ldsW[CC][136];  // W row-major: B[k][o] = ldsW[o][k]
  __shared__ _Float16 ldsA[16][136];

  const int tid = threadIdx.x;
  const int m0 = blockIdx.x * 16;
  const int b = blockIdx.y;
  const float* in_b = in + (size_t)b * CC * NN;
  float* out_b = out + (size_t)b * CC * NN;

  for (int i = tid; i < CC * CC; i += 256)
    ldsW[i >> 7][i & 127] = (_Float16)W[i];
  for (int i = tid; i < 16 * CC; i += 256) {
    int m = i >> 7, k = i & 127;
    float v = a_rowmajor ? in_b[(m0 + m) * CC + k] : in_b[k * NN + m0 + m];
    ldsA[m][k] = (_Float16)v;
  }
  __builtin_prefetch(bias ? (const void*)bias : (const void*)W, 0, 0);
  __syncthreads();

  const int wave = tid >> 5, lane = tid & 31;
  const int hi = lane >> 4, lm = lane & 15;
  const int o0 = wave * 16;

  v8f c = {};
  for (int kt = 0; kt < 4; ++kt) {
    const int kbA = kt * 32 + hi * 16;
    const int kbB = kt * 32 + hi * 16;
    v16h a, bf;
#pragma unroll
    for (int e = 0; e < 8; ++e) {
      a[e]     = ldsA[lm][kbA + e];
      a[8 + e] = ldsA[lm][kbA + 8 + e];
    }
#pragma unroll
    for (int e = 0; e < 16; ++e) bf[e] = ldsW[o0 + lm][kbB + e];
    c = wmma_f16(a, bf, c);
  }
  const int col = o0 + lm;
  const float bcol = bias ? bias[col] : 0.0f;
#pragma unroll
  for (int r = 0; r < 8; ++r) {
    const int row = m0 + r + hi * 8;
    out_b[row * CC + col] = c[r] * scale + bcol;
  }
}

// f32 -> f16 copies of Q and x (as (B,N,C) flat views)
__global__ void convert_kernel(const float* __restrict__ Qf,
                               const float* __restrict__ x,
                               _Float16* __restrict__ Qh,
                               _Float16* __restrict__ Xh, int n) {
  int i = blockIdx.x * blockDim.x + threadIdx.x;
  if (i < n) {
    Qh[i] = (_Float16)Qf[i];
    Xh[i] = (_Float16)x[i];
  }
}

// nq[row] = ||Q[row,:]||^2 ; one wave per row
__global__ __launch_bounds__(256) void rownorm_kernel(
    const float* __restrict__ Q, float* __restrict__ nq) {
  const int tid = threadIdx.x, wave = tid >> 5, lane = tid & 31;
  const int row = blockIdx.x * 8 + wave;
  const float* q = Q + (size_t)row * CC;
  float s = 0.0f;
  for (int t = lane; t < CC; t += 32) {
    float v = q[t];
    s += v * v;
  }
  s += __shfl_xor(s, 1, 32);
  s += __shfl_xor(s, 2, 32);
  s += __shfl_xor(s, 4, 32);
  s += __shfl_xor(s, 8, 32);
  s += __shfl_xor(s, 16, 32);
  if (lane == 0) nq[row] = s;
}

// Frobenius norms of Wq (block 0) and Wv (block 1)
__global__ __launch_bounds__(256) void frob_kernel(const float* __restrict__ Wq,
                                                   const float* __restrict__ Wv,
                                                   float* __restrict__ norms) {
  __shared__ float red[256];
  const float* W = blockIdx.x ? Wv : Wq;
  float s = 0.0f;
  for (int i = threadIdx.x; i < CC * CC; i += 256) s += W[i] * W[i];
  red[threadIdx.x] = s;
  __syncthreads();
  for (int off = 128; off; off >>= 1) {
    if ((int)threadIdx.x < off) red[threadIdx.x] += red[threadIdx.x + off];
    __syncthreads();
  }
  if (threadIdx.x == 0) norms[blockIdx.x] = sqrtf(red[0]);
}

// ---------------------------------------------------------------------------
// Fused L2-distance attention (flash-style, P never hits DRAM):
//   S[m,j] = 2*c1*(Q.Q^T)[m,j] - c1*(nq_m + nq_j);  P = softmax_j(S)
//   att[m,c] = sum_j P[m,j] * X[j,c]
// Block = 16 query rows; 4 waves split the 128 key-blocks (32 keys each).
// ---------------------------------------------------------------------------
__global__ __launch_bounds__(128) void attn_kernel(
    const _Float16* __restrict__ Qh, const _Float16* __restrict__ Xh,
    const float* __restrict__ nq, float* __restrict__ att) {
  __shared__ _Float16 ldsP[4][16][40];  // per-wave P tile: D-layout -> A-layout
  __shared__ float lacc[4][16][CC];     // per-wave partial accumulators
  __shared__ float lmax[4][16];
  __shared__ float lsum[4][16];

  const int tid = threadIdx.x, wave = tid >> 5, lane = tid & 31;
  const int hi = lane >> 4, lm = lane & 15;
  const int m0 = blockIdx.x * 16, b = blockIdx.y;
  const size_t bn = (size_t)b * NN;
  const _Float16* Qb = Qh + bn * CC;
  const _Float16* Xb = Xh + bn * CC;
  const float* nqb = nq + bn;
  const float c1 = 0.0883883476483184f;  // 1/sqrt(128)

  // Persistent A-fragments of the 16 query rows (K = 0..127)
  v16h aq[4];
#pragma unroll
  for (int kt = 0; kt < 4; ++kt) {
    const int kbA = kt * 32 + hi * 16;
    const _Float16* qm = Qb + (size_t)(m0 + lm) * CC;
#pragma unroll
    for (int e = 0; e < 8; ++e) {
      aq[kt][e]     = qm[kbA + e];
      aq[kt][8 + e] = qm[kbA + 8 + e];
    }
  }
  float nqm[8];
#pragma unroll
  for (int r = 0; r < 8; ++r) nqm[r] = nqb[m0 + r + hi * 8];

  float runmax[8], runsum[8], alpha[8], p0[8], p1[8];
  v8f zero = {};
  v8f acc[8];
#pragma unroll
  for (int r = 0; r < 8; ++r) {
    runmax[r] = -3.0e38f;
    runsum[r] = 0.0f;
  }
#pragma unroll
  for (int f = 0; f < 8; ++f) acc[f] = zero;

  for (int jb = wave; jb < NN / 32; jb += 4) {
    const int j0 = jb * 32;
    if (jb + 4 < NN / 32)
      __builtin_prefetch(Qb + (size_t)((jb + 4) * 32 + lm) * CC, 0, 0);

    const float nqj0 = nqb[j0 + lm];
    const float nqj1 = nqb[j0 + 16 + lm];

    // S tiles: S0 = rows x keys[j0..j0+15], S1 = keys[j0+16..j0+31]
    v8f s0f = zero, s1f = zero;
#pragma unroll
    for (int kt = 0; kt < 4; ++kt) {
      const int kbB = kt * 32 + hi * 16;
      const _Float16* qj0 = Qb + (size_t)(j0 + lm) * CC;
      const _Float16* qj1 = Qb + (size_t)(j0 + 16 + lm) * CC;
      v16h b0, b1;
#pragma unroll
      for (int e = 0; e < 16; ++e) {
        b0[e] = qj0[kbB + e];
        b1[e] = qj1[kbB + e];
      }
      s0f = wmma_f16(aq[kt], b0, s0f);
      s1f = wmma_f16(aq[kt], b1, s1f);
    }

    // Online softmax update (rows live on fixed 16-lane halves)
#pragma unroll
    for (int r = 0; r < 8; ++r) {
      const float s0 = 2.0f * c1 * s0f[r] - c1 * (nqm[r] + nqj0);
      const float s1 = 2.0f * c1 * s1f[r] - c1 * (nqm[r] + nqj1);
      const float tmax = redmax16(fmaxf(s0, s1));
      const float nm = fmaxf(runmax[r], tmax);
      alpha[r] = __expf(runmax[r] - nm);
      runmax[r] = nm;
      p0[r] = __expf(s0 - nm);
      p1[r] = __expf(s1 - nm);
      runsum[r] = runsum[r] * alpha[r] + redsum16(p0[r] + p1[r]);
    }
#pragma unroll
    for (int f = 0; f < 8; ++f)
#pragma unroll
      for (int r = 0; r < 8; ++r) acc[f][r] *= alpha[r];

    // P tile (16x32) through LDS: D-layout stores -> A-layout reads
#pragma unroll
    for (int r = 0; r < 8; ++r) {
      const int rr = r + hi * 8;
      ldsP[wave][rr][lm]      = (_Float16)p0[r];
      ldsP[wave][rr][16 + lm] = (_Float16)p1[r];
    }
    asm volatile("s_wait_dscnt 0" ::: "memory");
    v16h pA;
    const int kbP = hi * 16;
#pragma unroll
    for (int e = 0; e < 8; ++e) {
      pA[e]     = ldsP[wave][lm][kbP + e];
      pA[8 + e] = ldsP[wave][lm][kbP + 8 + e];
    }

    // acc += P(16x32) @ X(32x128)
#pragma unroll
    for (int f = 0; f < 8; ++f) {
      const int ccol = f * 16 + lm;
      const int kbB = hi * 16;
      v16h bx;
#pragma unroll
      for (int e = 0; e < 16; ++e)
        bx[e] = Xb[(size_t)(j0 + kbB + e) * CC + ccol];
      acc[f] = wmma_f16(pA, bx, acc[f]);
    }
  }

  // Merge the 4 per-wave flash states
  if (lm == 0) {
#pragma unroll
    for (int r = 0; r < 8; ++r) {
      lmax[wave][r + hi * 8] = runmax[r];
      lsum[wave][r + hi * 8] = runsum[r];
    }
  }
#pragma unroll
  for (int f = 0; f < 8; ++f)
#pragma unroll
    for (int r = 0; r < 8; ++r)
      lacc[wave][r + hi * 8][f * 16 + lm] = acc[f][r];
  __syncthreads();

  float* attb = att + bn * CC;
  for (int e = tid; e < 16 * CC; e += 128) {
    const int row = e >> 7, col = e & 127;
    float gm = -3.0e38f;
    for (int w = 0; w < 4; ++w) gm = fmaxf(gm, lmax[w][row]);
    float tot = 0.0f, val = 0.0f;
    for (int w = 0; w < 4; ++w) {
      const float fw = __expf(lmax[w][row] - gm);
      tot += lsum[w][row] * fw;
      val += lacc[w][row][col] * fw;
    }
    attb[(size_t)(m0 + row) * CC + col] = val / tot;
  }
}

__global__ void epilogue_kernel(const float* __restrict__ ob,
                                const float* __restrict__ x,
                                const float* __restrict__ gamma,
                                const float* __restrict__ norms, float phi,
                                float* __restrict__ out, int n) {
  int i = blockIdx.x * blockDim.x + threadIdx.x;
  if (i >= n) return;
  const float bound =
      sqrtf((float)NN / (float)CC) * (4.0f * phi + 1.0f) * norms[0] * norms[1];
  out[i] = gamma[0] * (ob[i] / bound) + x[i];
}

extern "C" void kernel_launch(void* const* d_in, const int* in_sizes, int n_in,
                              void* d_out, int out_size, void* d_ws,
                              size_t ws_size, hipStream_t stream) {
  (void)in_sizes; (void)n_in; (void)out_size; (void)ws_size;
  const float* x     = (const float*)d_in[0];
  const float* Wq    = (const float*)d_in[1];
  const float* bq    = (const float*)d_in[2];
  const float* Wv    = (const float*)d_in[3];
  const float* bv    = (const float*)d_in[4];
  const float* gamma = (const float*)d_in[5];
  float* out = (float*)d_out;

  const size_t tot = (size_t)BB * CC * NN;  // 2,097,152 elements
  char* ws = (char*)d_ws;
  float*    buf1  = (float*)ws;                            // 8 MB (Qf32 / att / att2)
  float*    buf2  = (float*)(ws + tot * 4);                // 8 MB (A_half / outbuf)
  _Float16* Qh    = (_Float16*)(ws + tot * 8);             // 4 MB
  _Float16* Xh    = (_Float16*)(ws + tot * 10);            // 4 MB
  float*    nqbuf = (float*)(ws + tot * 12);               // 64 KB
  float*    norms = (float*)(ws + tot * 12 + (size_t)BB * NN * 4);

  // Host-side Lambert W(N/e) (static constant, same Newton iteration as ref)
  double z = (double)NN / 2.718281828459045;
  double w = std::log(z) - std::log(std::log(z));
  for (int it = 0; it < 20; ++it) {
    double ew = std::exp(w);
    w = w - (w * ew - z) / (ew * (w + 1.0));
  }
  const float phi = (float)w;
  const float s2 = 1.0f / sqrtf((float)CC);

  dim3 ggrid(NN / 16, BB);
  // ||Wq||_F, ||Wv||_F
  frob_kernel<<<2, 256, 0, stream>>>(Wq, Wv, norms);
  // Q = conv1x1(x, Wq, bq)   (channel-major input view)
  conv_gemm_kernel<<<ggrid, 256, 0, stream>>>(x, Wq, bq, 1.0f, buf1, 0);
  // f16 copies of Q and x (as (B,N,C) views)
  convert_kernel<<<(int)((tot + 255) / 256), 256, 0, stream>>>(buf1, x, Qh, Xh,
                                                               (int)tot);
  // row norms ||q_n||^2
  rownorm_kernel<<<(BB * NN) / 8, 256, 0, stream>>>(buf1, nqbuf);
  // fused L2 softmax attention: att = softmax(-d2/sqrt(C)) @ x-view
  attn_kernel<<<ggrid, 128, 0, stream>>>(Qh, Xh, nqbuf, buf1);
  // A_half = conv1x1(att-view, Wq, bq)
  conv_gemm_kernel<<<ggrid, 256, 0, stream>>>(buf1, Wq, bq, 1.0f, buf2, 0);
  // att2 = s2 * A_half @ Wq^T   (row-major input view, no bias)
  conv_gemm_kernel<<<ggrid, 256, 0, stream>>>(buf2, Wq, nullptr, s2, buf1, 1);
  // out = conv1x1(att2-view, Wv, bv)
  conv_gemm_kernel<<<ggrid, 256, 0, stream>>>(buf1, Wv, bv, 1.0f, buf2, 0);
  // final: gamma * out / bound + x
  epilogue_kernel<<<(int)((tot + 255) / 256), 256, 0, stream>>>(
      buf2, x, gamma, norms, phi, out, (int)tot);
}
